// LanguageModel_50680614093410
// MI455X (gfx1250) — matrix operand order repaired and checked by reading, not compiled, not run on hardware
//
#include <hip/hip_runtime.h>

// ---------------------------------------------------------------------------
// Problem constants (from the reference)
// ---------------------------------------------------------------------------
static constexpr int VOC = 32000;
static constexpr int EMB = 512;
static constexpr int HID = 1024;
static constexpr int BB  = 16;     // batch
static constexpr int SS  = 256;    // sequence
static constexpr int G4  = 4 * HID;

// LSTM kernel geometry
static constexpr int NWG          = 32;           // persistent workgroups
static constexpr int LSTM_THREADS = 256;          // 8 waves
static constexpr int JPW          = HID / NWG;    // 32 hidden cols per wg

typedef __bf16 bf16_t;
typedef __attribute__((ext_vector_type(16))) __bf16 v16bf;
typedef __attribute__((ext_vector_type(8)))  __bf16 v8bf;
typedef __attribute__((ext_vector_type(8)))  float  v8f;

// ---------------------------------------------------------------------------
// WMMA helpers (CDNA5 16x16x32 bf16, f32 accumulate)
// ---------------------------------------------------------------------------
__device__ __forceinline__ v8f wmma_bf16(v16bf a, v16bf b, v8f c) {
    return __builtin_amdgcn_wmma_f32_16x16x32_bf16(
        /*neg_a=*/false, a, /*neg_b=*/false, b,
        /*c_mod=*/(short)0, c, /*reuse_a=*/false, /*reuse_b=*/false);
}

// A-matrix 16x32 bf16 tile, row-major source, row m = batch index.
// ISA layout: lanes 0-15 (M=lane): K=[kb..kb+7],[kb+16..kb+23];
//             lanes 16-31       : K=[kb+8..kb+15],[kb+24..kb+31].
__device__ __forceinline__ v16bf load_tile_A(const bf16_t* __restrict__ base,
                                             int ld, int kb) {
    const int lane = threadIdx.x & 31;
    const int m    = lane & 15;
    const int koff = (lane & 16) ? 8 : 0;
    const bf16_t* p = base + (size_t)m * ld + kb + koff;
    union { v16bf v; v8bf h[2]; } u;
    u.h[0] = *reinterpret_cast<const v8bf*>(p);
    u.h[1] = *reinterpret_cast<const v8bf*>(p + 16);
    return u.v;
}

// B-matrix 32x16 bf16 tile where B[k][n] = W[n][k], W row-major [N][K].
// lanes 0-15: col n=lane,   K=[kb..kb+7],[kb+16..kb+23]; lanes 16-31: K=[kb+8..kb+15],[kb+24..kb+31].
__device__ __forceinline__ v16bf load_tile_B(const bf16_t* __restrict__ W,
                                             int nbase, int ld, int kb) {
    const int lane = threadIdx.x & 31;
    const int n    = nbase + (lane & 15);
    const int k    = kb + ((lane & 16) ? 8 : 0);
    const bf16_t* p = W + (size_t)n * ld + k;
    union { v16bf v; v8bf h[2]; } u;
    u.h[0] = *reinterpret_cast<const v8bf*>(p);
    u.h[1] = *reinterpret_cast<const v8bf*>(p + 16);
    return u.v;
}

__device__ __forceinline__ float sigf(float x) {
    return 1.0f / (1.0f + __expf(-x));
}

// Async global->LDS 16-byte copy (per-lane), tracked by ASYNCcnt.
__device__ __forceinline__ void async_copy_b128(unsigned lds_off,
                                                const bf16_t* g) {
    asm volatile("global_load_async_to_lds_b128 %0, %1, off"
                 :: "v"(lds_off), "v"(g) : "memory");
}

// ---------------------------------------------------------------------------
// Prep kernels
// ---------------------------------------------------------------------------
__global__ void cast_f32_bf16(const float* __restrict__ src,
                              bf16_t* __restrict__ dst, int n) {
    int i = blockIdx.x * blockDim.x + threadIdx.x;
    if (i < n) dst[i] = (bf16_t)src[i];
}

__global__ void bias_combine(const float* __restrict__ a,
                             const float* __restrict__ b,
                             float* __restrict__ o, int n) {
    int i = blockIdx.x * blockDim.x + threadIdx.x;
    if (i < n) o[i] = a[i] + b[i];
}

// x_bf16[t][b][e] = bf16(emb[inputs[b][t]][e])
__global__ void embed_kernel(const int* __restrict__ idx,
                             const float* __restrict__ emb,
                             bf16_t* __restrict__ x, int n) {
    int i = blockIdx.x * blockDim.x + threadIdx.x;
    if (i >= n) return;
    int e  = i & (EMB - 1);
    int tb = i >> 9;            // EMB = 512 = 2^9
    int b  = tb & (BB - 1);
    int t  = tb >> 4;           // BB = 16 = 2^4
    int tok = idx[b * SS + t];
    x[i] = (bf16_t)emb[(size_t)tok * EMB + e];
}

__global__ void zero_u32(unsigned* __restrict__ p, int n) {
    int i = blockIdx.x * blockDim.x + threadIdx.x;
    if (i < n) p[i] = 0u;
}

// ---------------------------------------------------------------------------
// Persistent two-layer LSTM, software-pipelined: iteration p computes
// layer1 gates for t=p and layer2 gates for t=p-1 (both depend only on
// state published by the previous grid barrier).  One barrier per step.
// ---------------------------------------------------------------------------
__launch_bounds__(LSTM_THREADS, 1)
__global__ void lstm_kernel(const bf16_t* __restrict__ x,     // [S][B][EMB]
                            const bf16_t* __restrict__ W1ih,  // [4H][EMB]
                            const bf16_t* __restrict__ W1hh,  // [4H][HID]
                            const bf16_t* __restrict__ W2ih,  // [4H][HID]
                            const bf16_t* __restrict__ W2hh,  // [4H][HID]
                            const float*  __restrict__ bias1, // [4H]
                            const float*  __restrict__ bias2, // [4H]
                            bf16_t* __restrict__ h1buf,       // [2][B][HID]
                            bf16_t* __restrict__ h2seq,       // [S][B][HID]
                            unsigned* __restrict__ bar) {
    __shared__ float g1[4 * BB * JPW];   // layer1 gate tiles (f32)
    __shared__ float g2[4 * BB * JPW];   // layer2 gate tiles
    __shared__ float c1[BB * JPW];       // cell state, layer1 (this wg's slice)
    __shared__ float c2[BB * JPW];

    const int wg   = blockIdx.x;          // 0..NWG-1  -> hidden cols [wg*32, +32)
    const int tid  = threadIdx.x;
    const int w    = tid >> 5;            // wave 0..7
    const int lane = tid & 31;
    const int gate = w & 3;               // i,f,g,o
    const int jt   = w >> 2;              // 0..1 (16-wide j sub-tile)
    const int jl   = jt * 16 + (lane & 15);
    const int mb   = (lane & 16) ? 8 : 0;
    const int n1   = gate * HID + wg * JPW + jt * 16;  // weight row base

    for (int p = 0; p <= SS; ++p) {
        // -------- layer 1 gates, t = p
        if (p < SS) {
            v8f acc = {};
            const bf16_t* xa = x + (size_t)p * BB * EMB;
            for (int kb = 0; kb < EMB; kb += 32)
                acc = wmma_bf16(load_tile_A(xa, EMB, kb),
                                load_tile_B(W1ih, n1, EMB, kb), acc);
            if (p > 0) {
                const bf16_t* ha = h1buf + (size_t)((p - 1) & 1) * BB * HID;
                for (int kb = 0; kb < HID; kb += 32)
                    acc = wmma_bf16(load_tile_A(ha, HID, kb),
                                    load_tile_B(W1hh, n1, HID, kb), acc);
            }
#pragma unroll
            for (int j = 0; j < 8; ++j)
                g1[gate * (BB * JPW) + (mb + j) * JPW + jl] = acc[j];
        }
        // -------- layer 2 gates, t = p-1
        if (p >= 1) {
            v8f acc = {};
            const bf16_t* ha = h1buf + (size_t)((p - 1) & 1) * BB * HID;
            for (int kb = 0; kb < HID; kb += 32)
                acc = wmma_bf16(load_tile_A(ha, HID, kb),
                                load_tile_B(W2ih, n1, HID, kb), acc);
            if (p >= 2) {
                const bf16_t* h2a = h2seq + (size_t)(p - 2) * BB * HID;
                for (int kb = 0; kb < HID; kb += 32)
                    acc = wmma_bf16(load_tile_A(h2a, HID, kb),
                                    load_tile_B(W2hh, n1, HID, kb), acc);
            }
#pragma unroll
            for (int j = 0; j < 8; ++j)
                g2[gate * (BB * JPW) + (mb + j) * JPW + jl] = acc[j];
        }
        __syncthreads();

        // -------- elementwise cell updates (f32)
        if (p < SS) {
            for (int e = tid; e < BB * JPW; e += LSTM_THREADS) {
                int b = e / JPW, j = e % JPW, jg = wg * JPW + j;
                float iv = g1[0 * (BB * JPW) + e] + bias1[0 * HID + jg];
                float fv = g1[1 * (BB * JPW) + e] + bias1[1 * HID + jg];
                float gv = g1[2 * (BB * JPW) + e] + bias1[2 * HID + jg];
                float ov = g1[3 * (BB * JPW) + e] + bias1[3 * HID + jg];
                float c  = (p == 0) ? 0.0f : c1[e];
                float cn = sigf(fv) * c + sigf(iv) * tanhf(gv);
                c1[e] = cn;
                h1buf[(size_t)(p & 1) * BB * HID + (size_t)b * HID + jg] =
                    (bf16_t)(sigf(ov) * tanhf(cn));
            }
        }
        if (p >= 1) {
            for (int e = tid; e < BB * JPW; e += LSTM_THREADS) {
                int b = e / JPW, j = e % JPW, jg = wg * JPW + j;
                float iv = g2[0 * (BB * JPW) + e] + bias2[0 * HID + jg];
                float fv = g2[1 * (BB * JPW) + e] + bias2[1 * HID + jg];
                float gv = g2[2 * (BB * JPW) + e] + bias2[2 * HID + jg];
                float ov = g2[3 * (BB * JPW) + e] + bias2[3 * HID + jg];
                float c  = (p == 1) ? 0.0f : c2[e];
                float cn = sigf(fv) * c + sigf(iv) * tanhf(gv);
                c2[e] = cn;
                h2seq[(size_t)(p - 1) * BB * HID + (size_t)b * HID + jg] =
                    (bf16_t)(sigf(ov) * tanhf(cn));
            }
        }

        // -------- grid-wide barrier (monotone counter, sense by target)
        if (p < SS) {
            __threadfence();
            __syncthreads();
            if (tid == 0) {
                atomicAdd(bar, 1u);
                const unsigned target = (unsigned)(p + 1) * NWG;
                while (*(volatile unsigned*)bar < target)
                    __builtin_amdgcn_s_sleep(2);
            }
            __syncthreads();
            __threadfence();
        } else {
            __syncthreads();
        }
    }
}

// ---------------------------------------------------------------------------
// Logits GEMM: out[(b*S+t)][n] = H[t*16+b][:] . Wlog[n][:] + blog[n]
// Block tile 256(M) x 256(N); 16 waves in a 4x4 grid; each wave owns a
// 64x64 register tile (16 WMMA accumulators).  Per K-step the 256x32 A and
// B strips are staged into LDS once via async global->LDS DMA (ASYNCcnt),
// double-buffered against the WMMA consumption.
// ---------------------------------------------------------------------------
static constexpr int LSTRIDE = 40;   // bf16 elems per staged row (32 + 8 pad)

__launch_bounds__(512, 1)
__global__ void logits_kernel(const bf16_t* __restrict__ H,    // [4096][HID]
                              const bf16_t* __restrict__ Wl,   // [VOC][HID]
                              const float*  __restrict__ blog, // [VOC]
                              float* __restrict__ out) {       // [B][S][VOC]
    __shared__ bf16_t sA[2][256 * LSTRIDE];
    __shared__ bf16_t sB[2][256 * LSTRIDE];

    const int tid  = threadIdx.x;
    const int w    = tid >> 5;
    const int lane = tid & 31;
    const int mw   = w & 3;
    const int nw   = w >> 2;
    const int m0g  = blockIdx.y * 256;
    const int n0g  = blockIdx.x * 256;

    // 1024 16-byte chunks per strip; 2 chunks per thread per strip.
    const int ch0 = tid * 2;

    auto issue = [&](int buf, int kb) {
#pragma unroll
        for (int q = 0; q < 2; ++q) {
            int ch  = ch0 + q;          // 0..1023
            int row = ch >> 2;
            int c   = ch & 3;
            unsigned la = (unsigned)(size_t)&sA[buf][row * LSTRIDE + c * 8];
            async_copy_b128(la, H + (size_t)(m0g + row) * HID + kb + c * 8);
            unsigned lb = (unsigned)(size_t)&sB[buf][row * LSTRIDE + c * 8];
            async_copy_b128(lb, Wl + (size_t)(n0g + row) * HID + kb + c * 8);
        }
    };

    auto ldsA_tile = [&](int buf, int mt) -> v16bf {
        int m    = mw * 64 + mt * 16 + (lane & 15);
        int koff = (lane & 16) ? 8 : 0;
        const bf16_t* p = &sA[buf][m * LSTRIDE + koff];
        union { v16bf v; v8bf h[2]; } u;
        u.h[0] = *reinterpret_cast<const v8bf*>(p);
        u.h[1] = *reinterpret_cast<const v8bf*>(p + 16);
        return u.v;
    };
    auto ldsB_tile = [&](int buf, int nt) -> v16bf {
        int n = nw * 64 + nt * 16 + (lane & 15);
        int k = (lane & 16) ? 8 : 0;
        const bf16_t* p = &sB[buf][n * LSTRIDE + k];
        union { v16bf v; v8bf h[2]; } u;
        u.h[0] = *reinterpret_cast<const v8bf*>(p);
        u.h[1] = *reinterpret_cast<const v8bf*>(p + 16);
        return u.v;
    };

    v8f acc[4][4] = {};

    issue(0, 0);
    const int KSTEPS = HID / 32;
    for (int s = 0; s < KSTEPS; ++s) {
        const int buf = s & 1;
        if (s + 1 < KSTEPS) {
            issue(buf ^ 1, (s + 1) * 32);
            asm volatile("s_wait_asynccnt 0x4" ::: "memory");
        } else {
            asm volatile("s_wait_asynccnt 0x0" ::: "memory");
        }
        __syncthreads();                 // staged data visible to all waves

        v16bf a[4], b[4];
#pragma unroll
        for (int i = 0; i < 4; ++i) a[i] = ldsA_tile(buf, i);
#pragma unroll
        for (int i = 0; i < 4; ++i) b[i] = ldsB_tile(buf, i);
#pragma unroll
        for (int mi = 0; mi < 4; ++mi)
#pragma unroll
            for (int ni = 0; ni < 4; ++ni)
                acc[mi][ni] = wmma_bf16(a[mi], b[ni], acc[mi][ni]);

        __syncthreads();                 // protect buf from next-step overwrite
    }

    const int moff = (lane & 16) ? 8 : 0;
#pragma unroll
    for (int ni = 0; ni < 4; ++ni) {
        const int n  = n0g + nw * 64 + ni * 16 + (lane & 15);
        const float bv = blog[n];
#pragma unroll
        for (int mi = 0; mi < 4; ++mi) {
#pragma unroll
            for (int j = 0; j < 8; ++j) {
                int m = m0g + mw * 64 + mi * 16 + moff + j;
                int t = m >> 4, b = m & 15;
                out[(size_t)(b * SS + t) * VOC + n] = acc[mi][ni][j] + bv;
            }
        }
    }
}

// ---------------------------------------------------------------------------
// In-place log_softmax over rows of 32000
// ---------------------------------------------------------------------------
__launch_bounds__(256)
__global__ void logsoftmax_kernel(float* __restrict__ out) {
    __shared__ float red[256];
    float* p = out + (size_t)blockIdx.x * VOC;
    const int tid = threadIdx.x;

    float mx = -INFINITY;
    for (int i = tid; i < VOC; i += 256) mx = fmaxf(mx, p[i]);
    red[tid] = mx;
    __syncthreads();
    for (int s = 128; s > 0; s >>= 1) {
        if (tid < s) red[tid] = fmaxf(red[tid], red[tid + s]);
        __syncthreads();
    }
    mx = red[0];
    __syncthreads();

    float sum = 0.0f;
    for (int i = tid; i < VOC; i += 256) sum += __expf(p[i] - mx);
    red[tid] = sum;
    __syncthreads();
    for (int s = 128; s > 0; s >>= 1) {
        if (tid < s) red[tid] += red[tid + s];
        __syncthreads();
    }
    const float lse = mx + __logf(red[0]);
    __syncthreads();

    for (int i = tid; i < VOC; i += 256) p[i] = p[i] - lse;
}

// ---------------------------------------------------------------------------
// Launch
// ---------------------------------------------------------------------------
static inline size_t align256(size_t x) { return (x + 255) & ~(size_t)255; }

extern "C" void kernel_launch(void* const* d_in, const int* in_sizes, int n_in,
                              void* d_out, int out_size, void* d_ws, size_t ws_size,
                              hipStream_t stream) {
    (void)in_sizes; (void)n_in; (void)out_size; (void)ws_size;

    const int*   inputs = (const int*)  d_in[0];
    const float* emb    = (const float*)d_in[1];
    const float* W_ih1  = (const float*)d_in[2];
    const float* W_hh1  = (const float*)d_in[3];
    const float* b_ih1  = (const float*)d_in[4];
    const float* b_hh1  = (const float*)d_in[5];
    const float* W_ih2  = (const float*)d_in[6];
    const float* W_hh2  = (const float*)d_in[7];
    const float* b_ih2  = (const float*)d_in[8];
    const float* b_hh2  = (const float*)d_in[9];
    const float* W_log  = (const float*)d_in[10];
    const float* b_log  = (const float*)d_in[11];
    float* out = (float*)d_out;

    // Workspace layout (bytes, 256-aligned)
    char* ws = (char*)d_ws;
    size_t o = 0;
    bf16_t* w1ih = (bf16_t*)(ws + o); o = align256(o + (size_t)G4 * EMB * 2);
    bf16_t* w1hh = (bf16_t*)(ws + o); o = align256(o + (size_t)G4 * HID * 2);
    bf16_t* w2ih = (bf16_t*)(ws + o); o = align256(o + (size_t)G4 * HID * 2);
    bf16_t* w2hh = (bf16_t*)(ws + o); o = align256(o + (size_t)G4 * HID * 2);
    bf16_t* wlog = (bf16_t*)(ws + o); o = align256(o + (size_t)VOC * HID * 2);
    bf16_t* xbf  = (bf16_t*)(ws + o); o = align256(o + (size_t)SS * BB * EMB * 2);
    bf16_t* h1b  = (bf16_t*)(ws + o); o = align256(o + (size_t)2 * BB * HID * 2);
    bf16_t* h2s  = (bf16_t*)(ws + o); o = align256(o + (size_t)SS * BB * HID * 2);
    float*  bi1  = (float*) (ws + o); o = align256(o + (size_t)G4 * 4);
    float*  bi2  = (float*) (ws + o); o = align256(o + (size_t)G4 * 4);
    unsigned* cnt = (unsigned*)(ws + o); o = align256(o + 256);

    const int T = 256;
    auto blocks = [](size_t n, int t) { return (unsigned)((n + t - 1) / t); };

    // --- prep: casts, bias sums, embedding gather, barrier reset
    cast_f32_bf16<<<blocks((size_t)G4 * EMB, T), T, 0, stream>>>(W_ih1, w1ih, G4 * EMB);
    cast_f32_bf16<<<blocks((size_t)G4 * HID, T), T, 0, stream>>>(W_hh1, w1hh, G4 * HID);
    cast_f32_bf16<<<blocks((size_t)G4 * HID, T), T, 0, stream>>>(W_ih2, w2ih, G4 * HID);
    cast_f32_bf16<<<blocks((size_t)G4 * HID, T), T, 0, stream>>>(W_hh2, w2hh, G4 * HID);
    cast_f32_bf16<<<blocks((size_t)VOC * HID, T), T, 0, stream>>>(W_log, wlog, VOC * HID);
    bias_combine<<<blocks(G4, T), T, 0, stream>>>(b_ih1, b_hh1, bi1, G4);
    bias_combine<<<blocks(G4, T), T, 0, stream>>>(b_ih2, b_hh2, bi2, G4);
    embed_kernel<<<blocks((size_t)SS * BB * EMB, T), T, 0, stream>>>(
        inputs, emb, xbf, SS * BB * EMB);
    zero_u32<<<1, 64, 0, stream>>>(cnt, 64);

    // --- recurrent LSTM (persistent grid, one global barrier per step)
    lstm_kernel<<<NWG, LSTM_THREADS, 0, stream>>>(
        xbf, w1ih, w1hh, w2ih, w2hh, bi1, bi2, h1b, h2s, cnt);

    // --- logits GEMM: M=4096, N=32000, K=1024 (bf16 WMMA, async-LDS staged)
    dim3 ggrid(VOC / 256, (SS * BB) / 256);
    logits_kernel<<<ggrid, 512, 0, stream>>>(h2s, wlog, b_log, out);

    // --- log_softmax in place
    logsoftmax_kernel<<<SS * BB, 256, 0, stream>>>(out);
}
